// AdditiveAttention_71098888618436
// MI455X (gfx1250) — matrix operand order repaired and checked by reading, not compiled, not run on hardware
//
#include <hip/hip_runtime.h>
#include <hip/hip_bf16.h>
#include <math.h>

// ---------------------------------------------------------------------------
// AdditiveAttention fused kernel for gfx1250 (MI455X)
//   scores[i] = sum_j v[j] * tanh( sum_k x[i,k] * W[j,k] ),  x = [s | h]
//   out      = softmax(scores)
// GEMM runs on v_wmma_f32_16x16x32_bf16 (fp32 accumulate), X is staged in LDS as bf16.
// ---------------------------------------------------------------------------

typedef __attribute__((ext_vector_type(16))) __bf16 v16bf;
typedef __attribute__((ext_vector_type(8)))  __bf16 v8bf;
typedef __attribute__((ext_vector_type(8)))  float  v8f;

static constexpr int kD     = 1024;        // D == D_ATTN
static constexpr int kD2    = 2048;        // 2*D (concat width == K == N of GEMM)
static constexpr int kN     = 32768;       // rows
static constexpr int kRows  = 32;          // rows per workgroup (two 16-row M blocks)
static constexpr int kLDA   = kD2 + 8;     // LDS row stride in bf16 (16B pad -> bank skew)
static constexpr int kWaves = 8;           // 256 threads / wave32

// -------------------------------- W fp32 -> bf16 ---------------------------
__global__ __launch_bounds__(256) void wcvt_kernel(const float* __restrict__ W,
                                                   __bf16* __restrict__ Wb) {
  int i = (blockIdx.x * 256 + threadIdx.x) * 4;   // kD2*kD2 divisible by 1024
  float4 f = *(const float4*)(W + i);
  Wb[i + 0] = (__bf16)f.x;
  Wb[i + 1] = (__bf16)f.y;
  Wb[i + 2] = (__bf16)f.z;
  Wb[i + 3] = (__bf16)f.w;
}

// Build a 16-element bf16 A/B fragment from two contiguous 16-byte chunks.
__device__ inline v16bf join8(const __bf16* p0, const __bf16* p1) {
  union { v16bf v; v8bf h[2]; } u;
  u.h[0] = *(const v8bf*)p0;
  u.h[1] = *(const v8bf*)p1;
  return u.v;
}

// Branch-free tanh on the TRANS pipe: tanh(x) = (e^2x - 1) / (e^2x + 1).
// v_exp_f32 computes 2^x, so scale by 2*log2(e). Clamp avoids inf/inf.
__device__ inline float fast_tanh(float x) {
  const float xc = fminf(fmaxf(x, -16.f), 16.f);
  const float t  = __builtin_amdgcn_exp2f(xc * 2.885390081777927f); // e^(2x)
  return (t - 1.0f) * __builtin_amdgcn_rcpf(t + 1.0f);
}

// ------------------------- fused GEMM + tanh + v-dot -----------------------
__global__ __launch_bounds__(256) void attn_gemm_kernel(
    const float* __restrict__ S, const float* __restrict__ H,
    const __bf16* __restrict__ Wb, const float* __restrict__ V,
    float* __restrict__ scores)
{
  __shared__ __bf16 As[kRows * kLDA];          // 32 rows of x, bf16, padded
  __shared__ float  part[kWaves][kRows];       // per-wave row partial scores

  const int tid    = threadIdx.x;
  const int wave   = tid >> 5;
  const int lane   = tid & 31;
  const int laneM  = lane & 15;                // M (A) / N-column (B,C) index
  const int laneHi = lane >> 4;                // which 16-lane half
  const int row0   = blockIdx.x * kRows;

  // Stage x = concat(s, h) rows into LDS as bf16 (each element read once).
  for (int idx = tid; idx < kRows * kD2; idx += 256) {
    const int r = idx >> 11;                   // idx / 2048
    const int c = idx & (kD2 - 1);
    const float val = (c < kD) ? S[(row0 + r) * kD + c]
                               : H[(row0 + r) * kD + (c - kD)];
    As[r * kLDA + c] = (__bf16)val;
  }
  __syncthreads();

  float acc0[8], acc1[8];
  #pragma unroll
  for (int r = 0; r < 8; ++r) { acc0[r] = 0.f; acc1[r] = 0.f; }

  // A-fragment base pointers: 16-bit A layout (ISA 7.12.2):
  // lane holds row M=laneM, K chunks {base..base+7} and {base+16..base+23},
  // base = kk + 8*laneHi.
  const __bf16* A0base = &As[laneM * kLDA];          // M block 0 (rows 0..15)
  const __bf16* A1base = &As[(16 + laneM) * kLDA];   // M block 1 (rows 16..31)

  for (int jt = wave; jt < kD2 / 16; jt += kWaves) { // 128 j-tiles over 8 waves
    // B = W^T column j == W row j (K-contiguous). Lane holds column
    // N=laneM, K chunks {base..base+7} and {base+16..base+23}, base = kk + 8*laneHi.
    const __bf16* wrow = Wb + (size_t)(jt * 16 + laneM) * kD2;

    v8f cA0 = {0.f, 0.f, 0.f, 0.f, 0.f, 0.f, 0.f, 0.f};
    v8f cA1 = cA0;

    #pragma unroll 4
    for (int kk = 0; kk < kD2; kk += 32) {
      const v16bf b  = join8(wrow + kk + laneHi * 8, wrow + kk + laneHi * 8 + 16);   // 2x global b128 (L2-hot)
      const int   ca = kk + laneHi * 8;
      const v16bf a0 = join8(A0base + ca, A0base + ca + 16); // 2x ds_load_b128
      const v16bf a1 = join8(A1base + ca, A1base + ca + 16);
      cA0 = __builtin_amdgcn_wmma_f32_16x16x32_bf16(false, a0, false, b,
                                                    (short)0, cA0, false, false);
      cA1 = __builtin_amdgcn_wmma_f32_16x16x32_bf16(false, a1, false, b,
                                                    (short)0, cA1, false, false);
    }

    // C layout: VGPR r, lanes 0-15 -> M=r, lanes 16-31 -> M=8+r; N=laneM.
    const float vj = V[jt * 16 + laneM];
    #pragma unroll
    for (int r = 0; r < 8; ++r) {
      acc0[r] += vj * fast_tanh(cA0[r]);
      acc1[r] += vj * fast_tanh(cA1[r]);
    }
  }

  // Deterministic butterfly reduction over N within each 16-lane half.
  #pragma unroll
  for (int m = 1; m <= 8; m <<= 1) {
    #pragma unroll
    for (int r = 0; r < 8; ++r) {
      acc0[r] += __shfl_xor(acc0[r], m, 32);
      acc1[r] += __shfl_xor(acc1[r], m, 32);
    }
  }
  if (laneM == 0) {       // lane 0: rows r / 16+r ; lane 16: rows 8+r / 24+r
    #pragma unroll
    for (int r = 0; r < 8; ++r) {
      part[wave][laneHi * 8 + r]      = acc0[r];
      part[wave][16 + laneHi * 8 + r] = acc1[r];
    }
  }
  __syncthreads();

  if (tid < kRows) {      // ordered cross-wave sum -> bit-stable replays
    float t = 0.f;
    #pragma unroll
    for (int w = 0; w < kWaves; ++w) t += part[w][tid];
    scores[row0 + tid] = t;
  }
}

// ------------------------------ softmax(1, N) ------------------------------
__global__ __launch_bounds__(1024) void softmax_kernel(const float* __restrict__ scores,
                                                       float* __restrict__ out, int n) {
  __shared__ float buf[1024];
  const int tid = threadIdx.x;

  float m = -3.402823466e38f;
  for (int i = tid; i < n; i += 1024) m = fmaxf(m, scores[i]);
  buf[tid] = m;
  __syncthreads();
  for (int s2 = 512; s2 > 0; s2 >>= 1) {
    if (tid < s2) buf[tid] = fmaxf(buf[tid], buf[tid + s2]);
    __syncthreads();
  }
  const float gmax = buf[0];
  __syncthreads();

  float sum = 0.f;
  for (int i = tid; i < n; i += 1024) sum += __expf(scores[i] - gmax);
  buf[tid] = sum;
  __syncthreads();
  for (int s2 = 512; s2 > 0; s2 >>= 1) {
    if (tid < s2) buf[tid] += buf[tid + s2];
    __syncthreads();
  }
  const float inv = 1.f / buf[0];

  for (int i = tid; i < n; i += 1024) out[i] = __expf(scores[i] - gmax) * inv;
}

// ------------------------------- entry point -------------------------------
extern "C" void kernel_launch(void* const* d_in, const int* in_sizes, int n_in,
                              void* d_out, int out_size, void* d_ws, size_t ws_size,
                              hipStream_t stream) {
  const float* S = (const float*)d_in[0];   // (N, D)
  const float* H = (const float*)d_in[1];   // (N, D)
  const float* W = (const float*)d_in[2];   // (2*D_ATTN, 2*D) row-major
  const float* V = (const float*)d_in[3];   // (1, 2*D_ATTN)
  float* out = (float*)d_out;               // (1, N)

  __bf16* Wb    = (__bf16*)d_ws;                                        // 8 MB
  float* scores = (float*)((char*)d_ws + (size_t)kD2 * kD2 * sizeof(__bf16));

  wcvt_kernel<<<(kD2 * kD2) / (256 * 4), 256, 0, stream>>>(W, Wb);
  attn_gemm_kernel<<<kN / kRows, 256, 0, stream>>>(S, H, Wb, V, scores);
  softmax_kernel<<<1, 1024, 0, stream>>>(scores, out, kN);
}
